// SD2_67886253080729
// MI455X (gfx1250) — matrix operand order repaired and checked by reading, not compiled, not run on hardware
//
#include <hip/hip_runtime.h>
#include <hip/hip_bf16.h>

// ---------------------------------------------------------------------------
// MI455X (gfx1250) implementation of the DDI GNN forward pass.
//
// Roofline: all tensors fit in the 192MB L2 -> compute bound on ~135 GFLOP of
// dense GEMM.  Feature transforms use v_wmma_f32_16x16x32_f16 with 2x2
// register blocking (4 WMMAs / K-step, 32x32 tile per wave).  Weights are
// pre-packed once per call into the WMMA B-register layout in f16, so each
// B fragment is a single contiguous 32B load per lane.  Activations load as
// fp32 (vectorized b128 by the compiler) and convert in-register.
// Message passing / segment softmax: wave-per-edge reductions + f32 atomics.
// ---------------------------------------------------------------------------

typedef _Float16 half16 __attribute__((ext_vector_type(16)));
typedef float    float8 __attribute__((ext_vector_type(8)));

#define NNODES 16384
#define NEDGES 65536
#define NG     512

static __device__ __forceinline__ float elu1f(float x) {
    return x > 0.0f ? x : (expf(x) - 1.0f);
}

static __device__ __forceinline__ float atomicMaxFloat(float* addr, float v) {
    if (v >= 0.0f)
        return __int_as_float(atomicMax((int*)addr, __float_as_int(v)));
    else
        return __uint_as_float(atomicMin((unsigned int*)addr, __float_as_uint(v)));
}

// ---------------------------------------------------------------------------
// Pack fp32 W[K,Nc] (row-major, ldw=Nc) into f16 WMMA B-operand layout:
//   packed[ ((colTile*ksteps + ks)*32 + lane)*16 + h ]
//     = W[ ks*32 + (lane>>4)*16 + h , colTile*16 + (lane&15) ]
// One thread per (tile, lane) writes 16 contiguous halves.
// ---------------------------------------------------------------------------
__global__ void pack_B_f16(const float* __restrict__ W, int ldw, int K, int Nc,
                           _Float16* __restrict__ out)
{
    int idx = blockIdx.x * blockDim.x + threadIdx.x;
    int ksteps = K >> 5;
    int ntiles = (Nc >> 4) * ksteps;
    if (idx >= ntiles * 32) return;
    int lane = idx & 31;
    int t    = idx >> 5;
    int ks   = t % ksteps;
    int tn   = t / ksteps;
    int col  = tn * 16 + (lane & 15);
    int k0   = ks * 32 + (lane >> 4) * 16;
    _Float16* o = out + (size_t)idx * 16;
#pragma unroll
    for (int h = 0; h < 16; ++h)
        o[h] = (_Float16)W[(size_t)(k0 + h) * ldw + col];
}

// ---------------------------------------------------------------------------
// WMMA GEMM: Out[M,Nc] = act(A[M,Ktot] @ Wslice + bias).
// A: fp32 row-major (lda).  B: pre-packed f16 (colTile0 selects the column
// slice, e.g. per attention head).  M%32==0, Nc%32==0, Ktot%32==0.
// One wave -> 32x32 tile (2x2 WMMA register blocking); 8 waves per block.
// ---------------------------------------------------------------------------
__global__ void __launch_bounds__(256)
gemm_wmma_f16(const float* __restrict__ A, int lda,
              const _Float16* __restrict__ Bp, int Ktot, int colTile0,
              const float* __restrict__ bias,
              float* __restrict__ Out, int ldo,
              int M, int Ncols, int act)
{
    const int lane   = threadIdx.x & 31;
    const int wave   = threadIdx.x >> 5;
    const int tilesN = Ncols >> 5;
    const int tile   = blockIdx.x * 8 + wave;
    const int total  = (M >> 5) * tilesN;
    if (tile >= total) return;                 // wave-uniform: EXEC stays full

    const int tm    = tile / tilesN;
    const int tn    = tile - tm * tilesN;
    const int group = lane >> 4;               // 0: lanes 0-15, 1: lanes 16-31
    const int l16   = lane & 15;
    const int row0  = (tm << 5) + l16;         // first A row panel this lane
    const int ksteps  = Ktot >> 5;
    const size_t bstr = (size_t)ksteps * 32 * 16;     // halves per col-tile
    const _Float16* bp0 = Bp + (size_t)(colTile0 + (tn << 1)) * bstr + (size_t)lane * 16;
    const _Float16* bp1 = bp0 + bstr;

    float8 z = {0.f,0.f,0.f,0.f,0.f,0.f,0.f,0.f};
    float8 acc00 = z, acc01 = z, acc10 = z, acc11 = z;

    for (int ks = 0; ks < ksteps; ++ks) {
        const float* ar0 = A + (size_t)row0 * lda + ks * 32;
        const float* ar1 = ar0 + (size_t)16 * lda;
        __builtin_prefetch(ar0 + 32, 0, 1);    // next K panel -> global_prefetch_b8
        half16 a0, a1;
        // A 16x32 f16 layout (contiguous, matches packed B): element s of
        // lanes 0-15 -> K = s; lanes 16-31 -> K = 16 + s.
#pragma unroll
        for (int v = 0; v < 8; ++v) {
            int kk = group * 16 + v * 2;
            a0[2*v]   = (_Float16)ar0[kk];
            a0[2*v+1] = (_Float16)ar0[kk + 1];
            a1[2*v]   = (_Float16)ar1[kk];
            a1[2*v+1] = (_Float16)ar1[kk + 1];
        }
        // B fragments: one contiguous 32B half16 load per lane.
        half16 b0 = *(const half16*)(bp0 + (size_t)ks * 512);
        half16 b1 = *(const half16*)(bp1 + (size_t)ks * 512);
        acc00 = __builtin_amdgcn_wmma_f32_16x16x32_f16(false, a0, false, b0, (short)0, acc00, false, false);
        acc01 = __builtin_amdgcn_wmma_f32_16x16x32_f16(false, a0, false, b1, (short)0, acc01, false, false);
        acc10 = __builtin_amdgcn_wmma_f32_16x16x32_f16(false, a1, false, b0, (short)0, acc10, false, false);
        acc11 = __builtin_amdgcn_wmma_f32_16x16x32_f16(false, a1, false, b1, (short)0, acc11, false, false);
    }

    // C/D layout: VGPR r -> M = r + 8*group within the 16-row panel, N = l16.
    const int colL0 = (tn << 5) + l16;
    const int colL1 = colL0 + 16;
    const float b0v = bias ? bias[colL0] : 0.f;
    const float b1v = bias ? bias[colL1] : 0.f;
#pragma unroll
    for (int r = 0; r < 8; ++r) {
        int m0 = (tm << 5) + r + group * 8;
        int m1 = m0 + 16;
        float v00 = acc00[r] + b0v, v01 = acc01[r] + b1v;
        float v10 = acc10[r] + b0v, v11 = acc11[r] + b1v;
        if (act) { v00 = elu1f(v00); v01 = elu1f(v01); v10 = elu1f(v10); v11 = elu1f(v11); }
        Out[(size_t)m0 * ldo + colL0] = v00;
        Out[(size_t)m0 * ldo + colL1] = v01;
        Out[(size_t)m1 * ldo + colL0] = v10;
        Out[(size_t)m1 * ldo + colL1] = v11;
    }
}

// ---------------------------------------------------------------------------
// Elementwise fill
// ---------------------------------------------------------------------------
__global__ void fill_f32(float* p, float v, int n) {
    int i = blockIdx.x * blockDim.x + threadIdx.x;
    if (i < n) p[i] = v;
}

// ---------------------------------------------------------------------------
// PyG graph-mode LayerNorm over each graph's 32 nodes x 128 feats, in place,
// with optional fused ELU.  One block (256 thr) per graph.
// ---------------------------------------------------------------------------
__global__ void __launch_bounds__(256)
graph_ln_elu(float* __restrict__ X, const float* __restrict__ w,
             const float* __restrict__ b, int act)
{
    __shared__ float red[256];
    const int g = blockIdx.x;
    float* Xg = X + (size_t)g * 32 * 128;      // 4096 elements per graph

    float s = 0.f;
    for (int i = threadIdx.x; i < 4096; i += 256) s += Xg[i];
    red[threadIdx.x] = s; __syncthreads();
    for (int off = 128; off > 0; off >>= 1) {
        if (threadIdx.x < off) red[threadIdx.x] += red[threadIdx.x + off];
        __syncthreads();
    }
    const float mean = red[0] * (1.0f / 4096.0f);
    __syncthreads();

    float s2 = 0.f;
    for (int i = threadIdx.x; i < 4096; i += 256) {
        float d = Xg[i] - mean; s2 += d * d;
    }
    red[threadIdx.x] = s2; __syncthreads();
    for (int off = 128; off > 0; off >>= 1) {
        if (threadIdx.x < off) red[threadIdx.x] += red[threadIdx.x + off];
        __syncthreads();
    }
    const float inv = rsqrtf(red[0] * (1.0f / 4096.0f) + 1e-5f);

    for (int i = threadIdx.x; i < 4096; i += 256) {
        int f = i & 127;
        float v = (Xg[i] - mean) * inv * w[f] + b[f];
        if (act) v = elu1f(v);
        Xg[i] = v;
    }
}

// ---------------------------------------------------------------------------
// TransformerConv per-head attention logits: one wave per edge.
// alpha = <q[dst], k[src]+e> / sqrt(128); atomicMax segment max per dst.
// ---------------------------------------------------------------------------
__global__ void __launch_bounds__(256)
edge_alpha(const float* __restrict__ q, const float* __restrict__ k,
           const float* __restrict__ eF, const int* __restrict__ src,
           const int* __restrict__ dst, float* __restrict__ s,
           float* __restrict__ mbuf, int E)
{
    int e = blockIdx.x * 8 + (threadIdx.x >> 5);
    if (e >= E) return;
    const int lane = threadIdx.x & 31;
    const int sN = src[e], dN = dst[e];
    const float* qp = q  + (size_t)dN * 128;
    const float* kp = k  + (size_t)sN * 128;
    const float* ep = eF + (size_t)e  * 128;
    float sum = 0.f;
#pragma unroll
    for (int t = 0; t < 4; ++t) {
        int d = lane + 32 * t;
        sum += qp[d] * (kp[d] + ep[d]);
    }
    for (int off = 16; off > 0; off >>= 1) sum += __shfl_down(sum, off, 32);
    if (lane == 0) {
        float val = sum * 0.08838834764831845f;   // 1/sqrt(128)
        s[e] = val;
        atomicMaxFloat(mbuf + dN, val);
    }
}

__global__ void edge_exp(float* __restrict__ s, const int* __restrict__ dst,
                         const float* __restrict__ mbuf, float* __restrict__ z, int E)
{
    int e = blockIdx.x * blockDim.x + threadIdx.x;
    if (e >= E) return;
    int dN = dst[e];
    float t = expf(s[e] - mbuf[dN]);
    s[e] = t;
    atomicAdd(z + dN, t);
}

// msg += (1/H) * softmax_coeff * (v[src] + e), scattered to dst rows.
__global__ void edge_msg(const float* __restrict__ s, const float* __restrict__ z,
                         const float* __restrict__ vF, const float* __restrict__ eF,
                         const int* __restrict__ src, const int* __restrict__ dst,
                         float* __restrict__ out, int E)
{
    int idx = blockIdx.x * blockDim.x + threadIdx.x;
    if (idx >= E * 128) return;
    int e = idx >> 7, c = idx & 127;
    int dN = dst[e];
    float coeff = s[e] / (z[dN] + 1e-16f) * 0.25f;   // 1/H, H=4
    float val = (vF[(size_t)src[e] * 128 + c] + eF[(size_t)e * 128 + c]) * coeff;
    atomicAdd(out + (size_t)dN * 128 + c, val);
}

// ---------------------------------------------------------------------------
// SAGPool readout
// ---------------------------------------------------------------------------
__global__ void scatter_add_rows(const float* __restrict__ X, const int* __restrict__ src,
                                 const int* __restrict__ dst, float* __restrict__ agg, int E)
{
    int idx = blockIdx.x * blockDim.x + threadIdx.x;
    if (idx >= E * 128) return;
    int e = idx >> 7, c = idx & 127;
    atomicAdd(agg + (size_t)dst[e] * 128 + c, X[(size_t)src[e] * 128 + c]);
}

__global__ void pool_score(const float* __restrict__ X, const float* __restrict__ agg,
                           const float* __restrict__ Wrel, const float* __restrict__ brel,
                           const float* __restrict__ Wroot, float* __restrict__ score, int Nn)
{
    int n = blockIdx.x * blockDim.x + threadIdx.x;
    if (n >= Nn) return;
    const float* xr = X   + (size_t)n * 128;
    const float* ar = agg + (size_t)n * 128;
    float s = brel[0];
    for (int c = 0; c < 128; ++c) s += ar[c] * Wrel[c] + xr[c] * Wroot[c];
    score[n] = s;
}

// per-graph softmax of scores + weighted global-add-pool.  Block = 128 thr.
__global__ void __launch_bounds__(128)
pool_emb(const float* __restrict__ X, const float* __restrict__ score,
         float* __restrict__ emb)
{
    __shared__ float p[32];
    const int g = blockIdx.x;
    if (threadIdx.x < 32) p[threadIdx.x] = score[g * 32 + threadIdx.x];
    __syncthreads();
    if (threadIdx.x == 0) {
        float m = -1e30f;
        for (int i = 0; i < 32; ++i) m = fmaxf(m, p[i]);
        float zz = 0.f;
        for (int i = 0; i < 32; ++i) { p[i] = expf(p[i] - m); zz += p[i]; }
        float inv = 1.f / (zz + 1e-16f);
        for (int i = 0; i < 32; ++i) p[i] *= inv;
    }
    __syncthreads();
    const int c = threadIdx.x;
    float s = 0.f;
    for (int nloc = 0; nloc < 32; ++nloc)
        s += X[((size_t)g * 32 + nloc) * 128 + c] * p[nloc];
    emb[(size_t)g * 128 + c] = s;
}

// L2-normalize 128-wide rows with arbitrary row stride (in place).
__global__ void __launch_bounds__(128)
row_l2norm(float* __restrict__ X, int ld)
{
    __shared__ float red[128];
    float* r = X + (size_t)blockIdx.x * ld;
    float v = r[threadIdx.x];
    red[threadIdx.x] = v * v; __syncthreads();
    for (int off = 64; off > 0; off >>= 1) {
        if (threadIdx.x < off) red[threadIdx.x] += red[threadIdx.x + off];
        __syncthreads();
    }
    float inv = 1.f / fmaxf(sqrtf(red[0]), 1e-12f);
    r[threadIdx.x] = v * inv;
}

// standard LN over last dim (128) with weight/bias, eps 1e-5.
__global__ void __launch_bounds__(128)
ln_rows(const float* __restrict__ X, float* __restrict__ Y,
        const float* __restrict__ w, const float* __restrict__ b)
{
    __shared__ float red[128];
    const float* x = X + (size_t)blockIdx.x * 128;
    float v = x[threadIdx.x];
    red[threadIdx.x] = v; __syncthreads();
    for (int off = 64; off > 0; off >>= 1) {
        if (threadIdx.x < off) red[threadIdx.x] += red[threadIdx.x + off];
        __syncthreads();
    }
    float mean = red[0] * (1.0f / 128.0f);
    __syncthreads();
    float d = v - mean;
    red[threadIdx.x] = d * d; __syncthreads();
    for (int off = 64; off > 0; off >>= 1) {
        if (threadIdx.x < off) red[threadIdx.x] += red[threadIdx.x + off];
        __syncthreads();
    }
    float inv = rsqrtf(red[0] * (1.0f / 128.0f) + 1e-5f);
    Y[(size_t)blockIdx.x * 128 + threadIdx.x] = d * inv * w[threadIdx.x] + b[threadIdx.x];
}

// ---------------------------------------------------------------------------
// Co-attention (B=512, n=3 tokens, 3 heads x 128)
// ---------------------------------------------------------------------------
__global__ void co_dots(const float* __restrict__ Q, const float* __restrict__ KV,
                        float* __restrict__ dots)
{
    int idx = blockIdx.x * blockDim.x + threadIdx.x;     // 512*3*3*3
    if (idx >= NG * 27) return;
    int b = idx / 27, rem = idx % 27;
    int h = rem / 9, ij = rem % 9, i = ij / 3, j = ij % 3;
    const float* q = Q  + ((size_t)b * 3 + i) * 384 + h * 128;
    const float* k = KV + ((size_t)b * 3 + j) * 768 + h * 128;
    float s = 0.f;
    for (int d = 0; d < 128; ++d) s += q[d] * k[d];
    dots[idx] = s * 0.08838834764831845f;                // 1/sqrt(128)
}

__global__ void co_attnout(const float* __restrict__ dots, const float* __restrict__ KV,
                           float* __restrict__ Out)
{
    int idx = blockIdx.x * blockDim.x + threadIdx.x;     // 512*3*384
    if (idx >= NG * 3 * 384) return;
    int d  = idx % 384;
    int bi = idx / 384;
    int i  = bi % 3, b = bi / 3;
    int h  = d / 128, dd = d % 128;
    const float* dp = dots + b * 27 + h * 9 + i * 3;
    float m  = fmaxf(dp[0], fmaxf(dp[1], dp[2]));
    float e0 = expf(dp[0] - m), e1 = expf(dp[1] - m), e2 = expf(dp[2] - m);
    float zi = 1.f / (e0 + e1 + e2);
    const float* v0 = KV + (size_t)b * 3 * 768 + 384 + h * 128 + dd;
    Out[((size_t)b * 3 + i) * 384 + d] = (e0 * v0[0] + e1 * v0[768] + e2 * v0[1536]) * zi;
}

// logits = (ha . ta^T).reshape(9) @ mergeW + b ; softmax over 2 classes.
__global__ void final_merge(const float* __restrict__ ha, const float* __restrict__ ta,
                            const float* __restrict__ Wm, const float* __restrict__ bm,
                            float* __restrict__ out)
{
    int b = blockIdx.x * blockDim.x + threadIdx.x;
    if (b >= NG) return;
    float zm[9];
    for (int i = 0; i < 3; ++i)
        for (int j = 0; j < 3; ++j) {
            const float* a = ha + ((size_t)b * 3 + i) * 128;
            const float* c = ta + ((size_t)b * 3 + j) * 128;
            float s = 0.f;
            for (int d = 0; d < 128; ++d) s += a[d] * c[d];
            zm[i * 3 + j] = s;
        }
    float l0 = bm[0], l1 = bm[1];
    for (int k2 = 0; k2 < 9; ++k2) { l0 += zm[k2] * Wm[k2 * 2]; l1 += zm[k2] * Wm[k2 * 2 + 1]; }
    float m = fmaxf(l0, l1);
    float e0 = expf(l0 - m), e1 = expf(l1 - m);
    float zi = 1.f / (e0 + e1);
    out[b * 2]     = e0 * zi;
    out[b * 2 + 1] = e1 * zi;
}

// ---------------------------------------------------------------------------
// Host orchestration
// ---------------------------------------------------------------------------
struct ConvP  { const float *Wq,*bq,*Wk,*bk,*Wv,*bv,*We,*Wskip,*bskip; };
struct ConvPk { const _Float16 *Wq,*Wk,*Wv,*We,*Wskip; };
struct BlockP { ConvP c1, c2; const float *ln1w,*ln1b,*ln2w,*ln2b,*up1W,*up1b,*up2W,*up2b; };
struct BlockPk{ ConvPk c1, c2; const _Float16 *up1W,*up2W; };

static inline int cdiv(int a, int b) { return (a + b - 1) / b; }

extern "C" void kernel_launch(void* const* d_in, const int* in_sizes, int n_in,
                              void* d_out, int out_size, void* d_ws, size_t ws_size,
                              hipStream_t stream)
{
    if (n_in < 113) return;   // expect 10 graph arrays + 103 param leaves

    // ---- graph inputs (setup_inputs insertion order) ----
    const float* xin_d[2]  = { (const float*)d_in[0], (const float*)d_in[5] };
    const int*   ei_d[2]   = { (const int*)d_in[1],  (const int*)d_in[6]  };
    const float* eain_d[2] = { (const float*)d_in[2], (const float*)d_in[7] };

    // ---- params: recursive insertion-order flatten of the params pytree ----
    int cur = 10;
    auto nxt = [&]() { return (const float*)d_in[cur++]; };
    const float* init_node_W = nxt(); const float* init_node_b = nxt();
    const float* init_edge_W = nxt(); const float* init_edge_b = nxt();
    const float* init_ln_w   = nxt(); const float* init_ln_b   = nxt();
    BlockP blk[3];
    for (int i = 0; i < 3; ++i) {
        auto conv = [&]() {
            ConvP p;
            p.Wq = nxt(); p.bq = nxt(); p.Wk = nxt(); p.bk = nxt();
            p.Wv = nxt(); p.bv = nxt(); p.We = nxt();
            p.Wskip = nxt(); p.bskip = nxt();
            return p;
        };
        blk[i].c1 = conv(); blk[i].c2 = conv();
        blk[i].ln1w = nxt(); blk[i].ln1b = nxt();
        blk[i].ln2w = nxt(); blk[i].ln2b = nxt();
        blk[i].up1W = nxt(); blk[i].up1b = nxt();
        blk[i].up2W = nxt(); blk[i].up2b = nxt();
    }
    const float *Wrel[3], *brel[3], *Wroot[3];
    for (int i = 0; i < 3; ++i) { Wrel[i] = nxt(); brel[i] = nxt(); Wroot[i] = nxt(); }
    const float* Wt[3];
    for (int i = 0; i < 3; ++i) Wt[i] = nxt();
    const float* co_lnw = nxt(); const float* co_lnb = nxt();
    const float* co_Wq  = nxt(); const float* co_Wkv = nxt(); const float* co_Wo = nxt();
    const float* mergeW = nxt(); const float* mergeb = nxt();

    // ---- workspace carve (all offsets 256B aligned) ----
    char* base = (char*)d_ws; size_t off = 0;
    auto allocb = [&](size_t bytes) {
        void* p = (void*)(base + off);
        off += (bytes + 255) & ~(size_t)255;
        return p;
    };
    auto alloc  = [&](size_t elems) { return (float*)allocb(elems * sizeof(float)); };

    float* xcur  = alloc((size_t)NNODES * 128);
    float* xnext = alloc((size_t)NNODES * 128);
    float* eacur = alloc((size_t)NEDGES * 64);
    float* eanxt = alloc((size_t)NEDGES * 64);
    float* qh    = alloc((size_t)NNODES * 128);
    float* kh    = alloc((size_t)NNODES * 128);
    float* vh    = alloc((size_t)NNODES * 128);
    float* eh    = alloc((size_t)NEDGES * 128);
    float* sE    = alloc(NEDGES);
    float* mN    = alloc(NNODES);
    float* zN    = alloc(NNODES);
    float* agg   = alloc((size_t)NNODES * 128);
    float* score = alloc(NNODES);
    float* emb   = alloc((size_t)NG * 128);
    float* rh    = alloc((size_t)NG * 384);
    float* rt    = alloc((size_t)NG * 384);
    float* lnh   = alloc((size_t)NG * 384);
    float* lnt   = alloc((size_t)NG * 384);
    float* qA    = alloc((size_t)NG * 3 * 384);
    float* qB    = alloc((size_t)NG * 3 * 384);
    float* kvA   = alloc((size_t)NG * 3 * 768);
    float* kvB   = alloc((size_t)NG * 3 * 768);
    float* dotsA = alloc((size_t)NG * 27);
    float* dotsB = alloc((size_t)NG * 27);
    float* attA  = alloc((size_t)NG * 3 * 384);
    float* attB  = alloc((size_t)NG * 3 * 384);
    float* haB   = alloc((size_t)NG * 3 * 128);
    float* taB   = alloc((size_t)NG * 3 * 128);

    // ---- pack all weight matrices into f16 WMMA-B layout (once per call) ----
    auto pack = [&](const float* W, int K, int Nc) {
        _Float16* dst = (_Float16*)allocb((size_t)K * Nc * sizeof(_Float16));
        int nthr = (Nc >> 4) * (K >> 5) * 32;
        pack_B_f16<<<cdiv(nthr, 256), 256, 0, stream>>>(W, Nc, K, Nc, dst);
        return (const _Float16*)dst;
    };
    const _Float16* p_init_node = pack(init_node_W, 128, 128);
    const _Float16* p_init_edge = pack(init_edge_W, 64, 64);
    BlockPk pblk[3];
    for (int i = 0; i < 3; ++i) {
        auto pconv = [&](const ConvP& c) {
            ConvPk p;
            p.Wq = pack(c.Wq, 128, 512); p.Wk = pack(c.Wk, 128, 512);
            p.Wv = pack(c.Wv, 128, 512); p.We = pack(c.We, 64, 512);
            p.Wskip = pack(c.Wskip, 128, 128);
            return p;
        };
        pblk[i].c1 = pconv(blk[i].c1);
        pblk[i].c2 = pconv(blk[i].c2);
        pblk[i].up1W = pack(blk[i].up1W, 64, 64);
        pblk[i].up2W = pack(blk[i].up2W, 64, 64);
    }
    const _Float16* p_Wt[3];
    for (int i = 0; i < 3; ++i) p_Wt[i] = pack(Wt[i], 128, 128);
    const _Float16* p_coWq  = pack(co_Wq, 128, 384);
    const _Float16* p_coWkv = pack(co_Wkv, 128, 768);
    const _Float16* p_coWo  = pack(co_Wo, 384, 128);
    if (off > ws_size) return;  // workspace too small; nothing sane to do

    auto gemm = [&](const float* A, int lda, const _Float16* Bp, int Ktot, int ct0,
                    const float* bias, float* Out, int ldo, int M, int Nc, int act) {
        int tiles  = (M / 32) * (Nc / 32);
        int blocks = cdiv(tiles, 8);
        gemm_wmma_f16<<<blocks, 256, 0, stream>>>(A, lda, Bp, Ktot, ct0, bias, Out, ldo, M, Nc, act);
    };
    auto fill = [&](float* p, float v, int n) {
        fill_f32<<<cdiv(n, 256), 256, 0, stream>>>(p, v, n);
    };

    // TransformerConv: xout = skip(x) + mean_h( segment_softmax attention )
    auto tconv = [&](const float* xinp, const float* eainp, const ConvP& cp,
                     const ConvPk& pk, float* xout, const int* src, const int* dst) {
        gemm(xinp, 128, pk.Wskip, 128, 0, cp.bskip, xout, 128, NNODES, 128, 0);
        for (int h = 0; h < 4; ++h) {
            // head h = column tiles [8h, 8h+8) of the packed 512-wide weights
            gemm(xinp, 128, pk.Wq, 128, h * 8, cp.bq + h * 128, qh, 128, NNODES, 128, 0);
            gemm(xinp, 128, pk.Wk, 128, h * 8, cp.bk + h * 128, kh, 128, NNODES, 128, 0);
            gemm(xinp, 128, pk.Wv, 128, h * 8, cp.bv + h * 128, vh, 128, NNODES, 128, 0);
            gemm(eainp, 64, pk.We,  64, h * 8, nullptr,         eh, 128, NEDGES, 128, 0);
            fill(mN, -1e30f, NNODES);
            fill(zN, 0.0f,   NNODES);
            edge_alpha<<<NEDGES / 8, 256, 0, stream>>>(qh, kh, eh, src, dst, sE, mN, NEDGES);
            edge_exp<<<cdiv(NEDGES, 256), 256, 0, stream>>>(sE, dst, mN, zN, NEDGES);
            edge_msg<<<cdiv(NEDGES * 128, 256), 256, 0, stream>>>(sE, zN, vh, eh, src, dst, xout, NEDGES);
        }
    };

    // ---- per-drug encoder + readouts ----
    for (int drug = 0; drug < 2; ++drug) {
        const int* src = ei_d[drug];
        const int* dst = ei_d[drug] + NEDGES;
        float* rep = (drug == 0) ? rh : rt;

        // init: x = elu(graph_ln(x@W + b)); ea = elu(ea@We + be)
        gemm(xin_d[drug], 128, p_init_node, 128, 0, init_node_b, xcur, 128, NNODES, 128, 0);
        graph_ln_elu<<<NG, 256, 0, stream>>>(xcur, init_ln_w, init_ln_b, 1);
        gemm(eain_d[drug], 64, p_init_edge, 64, 0, init_edge_b, eacur, 64, NEDGES, 64, 1);

        for (int i = 0; i < 3; ++i) {
            // conv1 (cin->cin) then graph LN + ELU
            tconv(xcur, eacur, blk[i].c1, pblk[i].c1, xnext, src, dst);
            graph_ln_elu<<<NG, 256, 0, stream>>>(xnext, blk[i].ln1w, blk[i].ln1b, 1);
            // edge up1 with ELU
            gemm(eacur, 64, pblk[i].up1W, 64, 0, blk[i].up1b, eanxt, 64, NEDGES, 64, 1);
            // conv2 (cin->cout) using updated edges
            tconv(xnext, eanxt, blk[i].c2, pblk[i].c2, xcur, src, dst);
            // edge up2 with ELU -> next block's edge features
            gemm(eanxt, 64, pblk[i].up2W, 64, 0, blk[i].up2b, eacur, 64, NEDGES, 64, 1);
            graph_ln_elu<<<NG, 256, 0, stream>>>(xcur, blk[i].ln2w, blk[i].ln2b, 1);

            // SAGPool readout -> rep[:, i, :]
            fill(agg, 0.0f, NNODES * 128);
            scatter_add_rows<<<cdiv(NEDGES * 128, 256), 256, 0, stream>>>(xcur, src, dst, agg, NEDGES);
            pool_score<<<cdiv(NNODES, 256), 256, 0, stream>>>(xcur, agg, Wrel[i], brel[i], Wroot[i], score, NNODES);
            pool_emb<<<NG, 128, 0, stream>>>(xcur, score, emb);
            float* repSlice = rep + i * 128;                    // [NG,128], row stride 384
            gemm(emb, 128, p_Wt[i], 128, 0, nullptr, repSlice, 384, NG, 128, 0);
            row_l2norm<<<NG, 128, 0, stream>>>(repSlice, 384);
        }
    }

    // ---- co-attention ----
    ln_rows<<<NG * 3, 128, 0, stream>>>(rh, lnh, co_lnw, co_lnb);
    ln_rows<<<NG * 3, 128, 0, stream>>>(rt, lnt, co_lnw, co_lnb);
    gemm(lnh, 128, p_coWq, 128, 0, nullptr, qA, 384, NG * 3, 384, 0);    // queries from rh
    gemm(lnt, 128, p_coWkv, 128, 0, nullptr, kvB, 768, NG * 3, 768, 0);  // k/v from rt
    gemm(lnt, 128, p_coWq, 128, 0, nullptr, qB, 384, NG * 3, 384, 0);    // queries from rt
    gemm(lnh, 128, p_coWkv, 128, 0, nullptr, kvA, 768, NG * 3, 768, 0);  // k/v from rh
    co_dots<<<cdiv(NG * 27, 256), 256, 0, stream>>>(qA, kvB, dotsA);
    co_attnout<<<cdiv(NG * 3 * 384, 256), 256, 0, stream>>>(dotsA, kvB, attA);
    co_dots<<<cdiv(NG * 27, 256), 256, 0, stream>>>(qB, kvA, dotsB);
    co_attnout<<<cdiv(NG * 3 * 384, 256), 256, 0, stream>>>(dotsB, kvA, attB);
    gemm(attA, 384, p_coWo, 384, 0, nullptr, haB, 128, NG * 3, 128, 0);
    gemm(attB, 384, p_coWo, 384, 0, nullptr, taB, 128, NG * 3, 128, 0);

    final_merge<<<cdiv(NG, 256), 256, 0, stream>>>(haB, taB, mergeW, mergeb, (float*)d_out);
}
